// PartGNN_49727131353530
// MI455X (gfx1250) — compile-verified, matrix operand order checked by reading, not run on hardware
//
#include <hip/hip_runtime.h>
#include <hip/hip_fp16.h>

typedef __attribute__((ext_vector_type(16))) _Float16 v16h;
typedef __attribute__((ext_vector_type(8)))  float    v8f;

#define N_NODES    100000
#define N_EDGES    1600000
#define IN_CH      128
#define HID        128
#define EMBED      256
#define NUM_GRAPHS 128

// ---------------------------------------------------------------- utilities

__global__ void zero_f32_kernel(float* __restrict__ p, int n) {
  int i = blockIdx.x * blockDim.x + threadIdx.x;
  if (i < n) p[i] = 0.0f;
}

__global__ void cvt_f32_to_f16_kernel(const float* __restrict__ s,
                                      _Float16* __restrict__ d, int n) {
  int i = blockIdx.x * blockDim.x + threadIdx.x;
  if (i < n) d[i] = (_Float16)s[i];
}

// out_f16[i] = f16(a[i] + b[i])   (h + aggregated neighbors -> WMMA A matrix)
__global__ void fuse_add_to_f16_kernel(const float* __restrict__ a,
                                       const float* __restrict__ b,
                                       _Float16* __restrict__ o, int n) {
  int i = blockIdx.x * blockDim.x + threadIdx.x;
  if (i < n) o[i] = (_Float16)(a[i] + b[i]);
}

// ------------------------------------------------- edge gather + scatter-add
// one thread per (edge, 4-channel chunk): float4 gather, 4 fp32 atomic adds
__global__ void scatter_add_kernel(const float* __restrict__ feat,
                                   const int* __restrict__ srcI,
                                   const int* __restrict__ dstI,
                                   float* __restrict__ agg) {
  int idx = blockIdx.x * blockDim.x + threadIdx.x;   // over E * 32
  if (idx >= N_EDGES * 32) return;
  int e  = idx >> 5;
  int c4 = idx & 31;
  int s = srcI[e];
  int d = dstI[e];
  float4 v = ((const float4*)(feat + (size_t)s * 128))[c4];
  float* p = agg + (size_t)d * 128 + (size_t)c4 * 4;
  atomicAdd(p + 0, v.x);
  atomicAdd(p + 1, v.y);
  atomicAdd(p + 2, v.z);
  atomicAdd(p + 3, v.w);
}

// ------------------------------------------------------------- WMMA helpers
// 16-bit A-matrix 16x32 (MxK), ISA 7.12.2: lanes 0-15 -> M=lane, K {0..7,16..23};
// lanes 16-31 -> M=lane-16, K {8..15,24..31}. Pairs packed per VGPR.
__device__ __forceinline__ v16h load_frag_rowmajor(const _Float16* __restrict__ base,
                                                   int ld, int row0, int k0, int lane) {
  int half = lane >> 4;
  int r    = lane & 15;
  const _Float16* p = base + (size_t)(row0 + r) * ld + k0 + half * 8;
  v16h f;
#pragma unroll
  for (int i = 0; i < 8; ++i) f[i] = p[i];          // -> global/ds *_b128
#pragma unroll
  for (int i = 0; i < 8; ++i) f[8 + i] = p[16 + i]; // second 128b beat
  return f;
}

// A in row-major [M,128] fp16; W staged TRANSPOSED in LDS as [N][128] so the
// B fragment (lane = column n, same K striping as A) is two contiguous 128b
// ds loads. One wave owns a 16-row tile, loops all N-tiles, K=4x32 chunks.
template <int NCOLS, bool OUT_F16>
__global__ void gemm_bias_relu_kernel(const _Float16* __restrict__ A,
                                      const _Float16* __restrict__ W, // [128,NCOLS] row-major
                                      const float* __restrict__ bias,
                                      void* __restrict__ out) {
  extern __shared__ _Float16 smem[];                // NCOLS*128 halves, W^T
  const int tid = threadIdx.x;
  for (int i = tid; i < 128 * NCOLS; i += blockDim.x) {
    int k = i / NCOLS, n = i % NCOLS;
    smem[n * 128 + k] = W[i];                       // transpose into LDS
  }
  __syncthreads();

  const int wave = tid >> 5;
  const int lane = tid & 31;
  const int tile = blockIdx.x * (blockDim.x >> 5) + wave;
  if (tile >= N_NODES / 16) return;                 // wave-uniform exit
  const int m0 = tile * 16;

  // A fragments for all 4 K-chunks (kept in VGPRs, 2x global_load_b128 each)
  v16h a0 = load_frag_rowmajor(A, 128, m0, 0, lane);
  v16h a1 = load_frag_rowmajor(A, 128, m0, 32, lane);
  v16h a2 = load_frag_rowmajor(A, 128, m0, 64, lane);
  v16h a3 = load_frag_rowmajor(A, 128, m0, 96, lane);

  const int n_in_tile = lane & 15;                  // D layout: lane = column
  const int mbase     = m0 + (lane >> 4) * 8;       // VGPR r -> row mbase+r

#pragma unroll
  for (int nt = 0; nt < NCOLS / 16; ++nt) {
    v8f acc = {};
    v16h b0 = load_frag_rowmajor(smem, 128, nt * 16, 0, lane);
    acc = __builtin_amdgcn_wmma_f32_16x16x32_f16(false, a0, false, b0, (short)0, acc, false, false);
    v16h b1 = load_frag_rowmajor(smem, 128, nt * 16, 32, lane);
    acc = __builtin_amdgcn_wmma_f32_16x16x32_f16(false, a1, false, b1, (short)0, acc, false, false);
    v16h b2 = load_frag_rowmajor(smem, 128, nt * 16, 64, lane);
    acc = __builtin_amdgcn_wmma_f32_16x16x32_f16(false, a2, false, b2, (short)0, acc, false, false);
    v16h b3 = load_frag_rowmajor(smem, 128, nt * 16, 96, lane);
    acc = __builtin_amdgcn_wmma_f32_16x16x32_f16(false, a3, false, b3, (short)0, acc, false, false);

    const float bv = bias[nt * 16 + n_in_tile];
#pragma unroll
    for (int i = 0; i < 8; ++i) {
      float v = fmaxf(acc[i] + bv, 0.0f);           // fused bias + ReLU
      size_t o = (size_t)(mbase + i) * NCOLS + nt * 16 + n_in_tile;
      if (OUT_F16) ((_Float16*)out)[o] = (_Float16)v;
      else         ((float*)out)[o]    = v;
    }
  }
}

// ----------------------------------------------- segmented mean pool (sorted)
__global__ void pool_mean_kernel(const float* __restrict__ h2,
                                 const int* __restrict__ batch,
                                 float* __restrict__ pooled) {
  const int g = blockIdx.x;
  __shared__ int bounds[2];
  if (threadIdx.x == 0) {
    int lo = 0, hi = N_NODES;                       // lower_bound(batch, g)
    while (lo < hi) { int m = (lo + hi) >> 1; if (batch[m] < g) lo = m + 1; else hi = m; }
    bounds[0] = lo;
    lo = 0; hi = N_NODES;                           // lower_bound(batch, g+1)
    while (lo < hi) { int m = (lo + hi) >> 1; if (batch[m] < g + 1) lo = m + 1; else hi = m; }
    bounds[1] = lo;
  }
  __syncthreads();
  const int s = bounds[0], e = bounds[1];
  const int c = threadIdx.x;                        // blockDim == EMBED
  float sum = 0.0f;
  for (int n = s; n < e; ++n) sum += h2[(size_t)n * EMBED + c];
  pooled[(size_t)g * EMBED + c] = sum / fmaxf((float)(e - s), 1.0f);
}

// --------------------------------------------------------------- layer norm
__global__ void layernorm_kernel(const float* __restrict__ pooled,
                                 const float* __restrict__ gamma,
                                 const float* __restrict__ beta,
                                 float* __restrict__ out) {
  const int g = blockIdx.x;
  const int c = threadIdx.x;                        // 256 threads = 8 waves
  float v = pooled[(size_t)g * EMBED + c];
  float s = v, s2 = v * v;
#pragma unroll
  for (int off = 16; off > 0; off >>= 1) {          // wave32 reduce
    s  += __shfl_xor(s, off, 32);
    s2 += __shfl_xor(s2, off, 32);
  }
  __shared__ float red[2][8];
  const int lane = c & 31, w = c >> 5;
  if (lane == 0) { red[0][w] = s; red[1][w] = s2; }
  __syncthreads();
  __shared__ float mv[2];
  if (c == 0) {
    float ts = 0.0f, ts2 = 0.0f;
#pragma unroll
    for (int i = 0; i < 8; ++i) { ts += red[0][i]; ts2 += red[1][i]; }
    float mean = ts / (float)EMBED;
    float var  = ts2 / (float)EMBED - mean * mean;  // population var (jnp.var)
    mv[0] = mean;
    mv[1] = rsqrtf(var + 1e-5f);
  }
  __syncthreads();
  out[(size_t)g * EMBED + c] = (v - mv[0]) * mv[1] * gamma[c] + beta[c];
}

// ------------------------------------------------------------------- driver
extern "C" void kernel_launch(void* const* d_in, const int* in_sizes, int n_in,
                              void* d_out, int out_size, void* d_ws, size_t ws_size,
                              hipStream_t stream) {
  const float* x     = (const float*)d_in[0];
  const int*   ei    = (const int*)d_in[1];     // [2, E] flat
  const int*   batch = (const int*)d_in[2];
  const float* W11f  = (const float*)d_in[3];
  const float* b11   = (const float*)d_in[4];
  const float* W12f  = (const float*)d_in[5];
  const float* b12   = (const float*)d_in[6];
  const float* W21f  = (const float*)d_in[7];
  const float* b21   = (const float*)d_in[8];
  const float* W22f  = (const float*)d_in[9];
  const float* b22   = (const float*)d_in[10];
  const float* gamma = (const float*)d_in[11];
  const float* beta  = (const float*)d_in[12];
  float* out = (float*)d_out;

  const int* srcI = ei;
  const int* dstI = ei + N_EDGES;

  // workspace carve-up (~154 MB); agg & h1 alias the later h2 region (both
  // dead before h2 is written by the last GEMM)
  char* ws = (char*)d_ws;
  size_t off = 0;
  auto carve = [&](size_t bytes) { char* p = ws + off; off += (bytes + 255) & ~(size_t)255; return p; };
  _Float16* B1     = (_Float16*)carve((size_t)N_NODES * 128 * sizeof(_Float16));
  _Float16* B2     = (_Float16*)carve((size_t)N_NODES * 128 * sizeof(_Float16));
  float*    H2     = (float*)   carve((size_t)N_NODES * EMBED * sizeof(float));
  _Float16* Wf     = (_Float16*)carve((size_t)(3 * 128 * 128 + 128 * EMBED) * sizeof(_Float16));
  float*    pooled = (float*)   carve((size_t)NUM_GRAPHS * EMBED * sizeof(float));
  (void)ws_size; (void)in_sizes; (void)n_in; (void)out_size;

  float* agg = H2;                                  // N x 128 f32
  float* h1  = H2 + (size_t)N_NODES * 128;          // N x 128 f32

  _Float16* W11h = Wf;
  _Float16* W12h = Wf + 128 * 128;
  _Float16* W21h = Wf + 2 * 128 * 128;
  _Float16* W22h = Wf + 3 * 128 * 128;

  const int NF   = N_NODES * 128;                   // 12.8M node-feature elems
  const int EB   = N_EDGES * 32;                    // edge work items (x4 ch)
  const int ZB   = (NF + 255) / 256;
  const int SB   = (EB + 255) / 256;
  const int GB   = (N_NODES / 16 + 3) / 4;          // 4 waves/block, 1 tile/wave

  // weights -> fp16
  cvt_f32_to_f16_kernel<<<(128 * 128 + 255) / 256, 256, 0, stream>>>(W11f, W11h, 128 * 128);
  cvt_f32_to_f16_kernel<<<(128 * 128 + 255) / 256, 256, 0, stream>>>(W12f, W12h, 128 * 128);
  cvt_f32_to_f16_kernel<<<(128 * 128 + 255) / 256, 256, 0, stream>>>(W21f, W21h, 128 * 128);
  cvt_f32_to_f16_kernel<<<(128 * EMBED + 255) / 256, 256, 0, stream>>>(W22f, W22h, 128 * EMBED);

  // ---- layer 1 ----
  zero_f32_kernel<<<ZB, 256, 0, stream>>>(agg, NF);
  scatter_add_kernel<<<SB, 256, 0, stream>>>(x, srcI, dstI, agg);
  fuse_add_to_f16_kernel<<<ZB, 256, 0, stream>>>(x, agg, B1, NF);
  gemm_bias_relu_kernel<128, true ><<<GB, 128, 128 * 128 * sizeof(_Float16), stream>>>(B1, W11h, b11, B2);
  gemm_bias_relu_kernel<128, false><<<GB, 128, 128 * 128 * sizeof(_Float16), stream>>>(B2, W12h, b12, h1);

  // ---- layer 2 ----
  zero_f32_kernel<<<ZB, 256, 0, stream>>>(agg, NF);
  scatter_add_kernel<<<SB, 256, 0, stream>>>(h1, srcI, dstI, agg);
  fuse_add_to_f16_kernel<<<ZB, 256, 0, stream>>>(h1, agg, B1, NF);
  gemm_bias_relu_kernel<128, true ><<<GB, 128, 128 * 128 * sizeof(_Float16), stream>>>(B1, W21h, b21, B2);
  gemm_bias_relu_kernel<EMBED, false><<<GB, 128, 128 * EMBED * sizeof(_Float16), stream>>>(B2, W22h, b22, H2);

  // ---- pool + layernorm ----
  pool_mean_kernel<<<NUM_GRAPHS, EMBED, 0, stream>>>(H2, batch, pooled);
  layernorm_kernel<<<NUM_GRAPHS, EMBED, 0, stream>>>(pooled, gamma, beta, out);
}
